// TTTModel_6597069766657
// MI455X (gfx1250) — matrix operand order repaired and checked by reading, not compiled, not run on hardware
//
#include <hip/hip_runtime.h>
#include <hip/hip_bf16.h>

typedef __attribute__((ext_vector_type(16))) __bf16 v16bf;
typedef __attribute__((ext_vector_type(8)))  float  v8f;
typedef __attribute__((ext_vector_type(2)))  float  v2f;

#define LRATE 0.1f
#define NWG   32          // scan workgroups; each owns 16 of H=512 columns

// ---------------- Tensor Data Mover support (probe-verified builtins) ----------
#if defined(__HIP_DEVICE_COMPILE__) && __has_builtin(__builtin_amdgcn_tensor_load_to_lds)
#define HAVE_TDM 1
#else
#define HAVE_TDM 0
#endif

#if HAVE_TDM
typedef __attribute__((ext_vector_type(4))) unsigned int v4u;
typedef __attribute__((ext_vector_type(8))) int          v8i;
typedef __attribute__((ext_vector_type(4))) int          v4i;

// Pack a 2-D (or 1-D when td1==0) tile descriptor per CDNA5 D# layout.
// dsz: 0=1B,1=2B,2=4B,3=8B.  stride in data_size units.
__device__ __forceinline__ void tdm_desc(v4u& g0, v8i& g1,
                                         unsigned lds, unsigned long long ga,
                                         unsigned td0, unsigned td1,
                                         unsigned nd0, unsigned nd1,
                                         unsigned long long stride, unsigned dsz){
  g0[0] = 1u;                                   // count=1, user descriptor
  g0[1] = lds;                                  // lds_addr
  g0[2] = (unsigned)(ga & 0xffffffffull);       // global_addr[31:0]
  g0[3] = (unsigned)((ga >> 32) & 0x1ffffffull) | (2u << 30);  // addr[56:32] | type=2
  g1[0] = (int)(dsz << 16);                     // wg_mask=0, data_size, no pad/iter
  g1[1] = (int)((nd0 & 0xffffu) << 16);         // atomic_addr=0 | tensor_dim0[15:0]
  g1[2] = (int)((nd0 >> 16) | ((nd1 & 0xffffu) << 16));  // dim0 hi | dim1 lo
  g1[3] = (int)((nd1 >> 16) | (td0 << 16));     // dim1 hi | tile_dim0
  g1[4] = (int)(td1 & 0xffffu);                 // tile_dim1 | tile_dim2=0
  g1[5] = (int)(stride & 0xffffffffull);        // tensor_dim0_stride[31:0]
  g1[6] = (int)((stride >> 32) & 0xffffull);    // stride hi | dim1_stride lo = 0
  g1[7] = 0;
}

__device__ __forceinline__ void tdm_load(unsigned lds, unsigned long long ga,
                                         unsigned td0, unsigned td1,
                                         unsigned nd0, unsigned nd1,
                                         unsigned long long stride, unsigned dsz){
  v4u g0; v8i g1; tdm_desc(g0, g1, lds, ga, td0, td1, nd0, nd1, stride, dsz);
  v4i z4 = {0,0,0,0};
#if __clang_major__ >= 23
  v8i z8 = {0,0,0,0,0,0,0,0};
  __builtin_amdgcn_tensor_load_to_lds(g0, g1, z4, z4, z8, 0);
#else
  __builtin_amdgcn_tensor_load_to_lds(g0, g1, z4, z4, 0);
#endif
}

#if __has_builtin(__builtin_amdgcn_tensor_store_from_lds)
#define HAVE_TDM_ST 1
__device__ __forceinline__ void tdm_store(unsigned lds, unsigned long long ga,
                                          unsigned td0, unsigned td1,
                                          unsigned nd0, unsigned nd1,
                                          unsigned long long stride, unsigned dsz){
  v4u g0; v8i g1; tdm_desc(g0, g1, lds, ga, td0, td1, nd0, nd1, stride, dsz);
  v4i z4 = {0,0,0,0};
#if __clang_major__ >= 23
  v8i z8 = {0,0,0,0,0,0,0,0};
  __builtin_amdgcn_tensor_store_from_lds(g0, g1, z4, z4, z8, 0);
#else
  __builtin_amdgcn_tensor_store_from_lds(g0, g1, z4, z4, 0);
#endif
}
#else
#define HAVE_TDM_ST 0
#endif

#define LDS_OFF(p) ((unsigned)(size_t)((void*)(p)))
#define GA(p)      ((unsigned long long)(size_t)((const void*)(p)))
#endif  // HAVE_TDM

// ---------------- math helpers (jax.nn.gelu tanh approximation) ----------------
__device__ __forceinline__ float gelu_f(float x){
  const float c0 = 0.7978845608028654f, c1 = 0.044715f;
  float t = tanhf(c0*(x + c1*x*x*x));
  return 0.5f*x*(1.0f+t);
}
__device__ __forceinline__ float gelu_grad_f(float x){
  const float c0 = 0.7978845608028654f, c1 = 0.044715f;
  float u = c0*(x + c1*x*x*x);
  float t = tanhf(u);
  float du = c0*(1.0f + 3.0f*c1*x*x);
  return 0.5f*(1.0f+t) + 0.5f*x*(1.0f-t*t)*du;
}

// ---------------- grid-wide barrier (32 resident WGs, generation counter) ------
__device__ __forceinline__ void gsync(unsigned* sync){
  __syncthreads();
  if (threadIdx.x == 0){
    __threadfence();
    unsigned gen = __hip_atomic_load(&sync[1], __ATOMIC_RELAXED, __HIP_MEMORY_SCOPE_AGENT);
    unsigned arr = __hip_atomic_fetch_add(&sync[0], 1u, __ATOMIC_ACQ_REL, __HIP_MEMORY_SCOPE_AGENT);
    if (arr == NWG-1){
      __hip_atomic_store(&sync[0], 0u, __ATOMIC_RELAXED, __HIP_MEMORY_SCOPE_AGENT);
      __hip_atomic_fetch_add(&sync[1], 1u, __ATOMIC_RELEASE, __HIP_MEMORY_SCOPE_AGENT);
    } else {
      while (__hip_atomic_load(&sync[1], __ATOMIC_ACQUIRE, __HIP_MEMORY_SCOPE_AGENT) == gen)
        __builtin_amdgcn_s_sleep(1);
    }
    __threadfence();
  }
  __syncthreads();
}

// ---------------- cast f32 -> bf16 ----------------
__global__ void cast_f32_to_bf16(const float* __restrict__ src,
                                 __hip_bfloat16* __restrict__ dst, int n){
  int i = blockIdx.x*blockDim.x + threadIdx.x;
  if (i < n) dst[i] = __float2bfloat16(src[i]);
}

// ---------------- per-launch state init (deterministic across graph replays) ---
__global__ void init_state(const float* __restrict__ W3,
                           float* __restrict__ W3g,
                           unsigned* __restrict__ sync){
  int i = blockIdx.x*blockDim.x + threadIdx.x;
  if (i < 64) sync[i] = 0u;
  for (int j = i; j < 512*512; j += gridDim.x*blockDim.x) W3g[j] = W3[j];
}

// ---------------- bf16 WMMA GEMM: C(MxN,f32) = A(MxK) * B(KxN), row-major ------
// TDM double-buffered pipeline: wave0 issues tensor_load_to_lds for chunk i+1
// while all waves run WMMAs on chunk i.
#define GT_M 64
#define GT_N 32
#define GT_K 32
__global__ __launch_bounds__(256)
void gemm_bf16_wmma(const __hip_bfloat16* __restrict__ Abf,
                    const __hip_bfloat16* __restrict__ Bbf,
                    float* __restrict__ C,
                    int M, int N, int K,
                    size_t strideB, size_t strideC){
  const unsigned short* A = reinterpret_cast<const unsigned short*>(Abf);
  const unsigned short* B = reinterpret_cast<const unsigned short*>(Bbf)
                          + (size_t)blockIdx.z * strideB;
  float* Cz = C + (size_t)blockIdx.z * strideC;

  __shared__ __attribute__((aligned(16))) unsigned short As[2][GT_M*GT_K];
  __shared__ __attribute__((aligned(16))) unsigned short Bs[2][GT_K*GT_N];

  const int tid  = threadIdx.x;
  const int wave = tid >> 5, lane = tid & 31;
  const int wm = wave >> 1, wn = wave & 1;   // 4x2 waves of 16x16 tiles
  const int bm = blockIdx.x * GT_M;
  const int bn = blockIdx.y * GT_N;
  const int r  = lane & 15, hi = lane >> 4;
  const int nk = K / GT_K;

  v8f acc = {};

#if HAVE_TDM
  if (wave == 0){   // uniform branch: exactly one wave issues the DMAs
    tdm_load(LDS_OFF(&As[0][0]), GA(A + (size_t)bm*K),       GT_K, GT_M,
             GT_K, GT_M, (unsigned long long)K, 1);
    tdm_load(LDS_OFF(&Bs[0][0]), GA(B + (size_t)bn),         GT_N, GT_K,
             GT_N, GT_K, (unsigned long long)N, 1);
  }
  int cur = 0;
  for (int ik = 0; ik < nk; ++ik){
    if (wave == 0) __builtin_amdgcn_s_wait_tensorcnt(0);
    __syncthreads();                       // chunk ik resident; chunk ik-1 consumed
    if (wave == 0 && ik + 1 < nk){
      const int k0n = (ik+1)*GT_K;
      tdm_load(LDS_OFF(&As[cur^1][0]), GA(A + (size_t)bm*K + k0n), GT_K, GT_M,
               GT_K, GT_M, (unsigned long long)K, 1);
      tdm_load(LDS_OFF(&Bs[cur^1][0]), GA(B + (size_t)k0n*N + bn), GT_N, GT_K,
               GT_N, GT_K, (unsigned long long)N, 1);
    }
    union { unsigned short u[16]; uint4 q[2]; v16bf v; } af, bf_;
    const int m0 = wm*16, n0 = wn*16;
    { // A frag: row m0+r; K = {8*hi+0..7} then {16+8*hi+0..7}
      const unsigned short* arow = &As[cur][(m0+r)*GT_K];
      af.q[0] = *(const uint4*)&arow[8*hi];
      af.q[1] = *(const uint4*)&arow[16 + 8*hi];
    }
    #pragma unroll
    for (int e = 0; e < 16; ++e)  // B frag: col n0+r; K = 8*hi + (e&7) + 16*(e>>3)
      bf_.u[e] = Bs[cur][(8*hi + (e&7) + 16*(e>>3))*GT_N + n0 + r];
    acc = __builtin_amdgcn_wmma_f32_16x16x32_bf16(false, af.v, false, bf_.v,
                                                  (short)0, acc, false, false);
    cur ^= 1;
  }
#else
  for (int ik = 0; ik < nk; ++ik){
    const int k0 = ik*GT_K;
    __syncthreads();
    { int idx = tid*8, ar = idx >> 5, ac = idx & 31;
      *(uint4*)&As[0][idx] = *(const uint4*)(A + (size_t)(bm+ar)*K + k0 + ac); }
    if (tid < 128){
      int idx = tid*8, br = idx >> 5, bc = idx & 31;
      *(uint4*)&Bs[0][idx] = *(const uint4*)(B + (size_t)(k0+br)*N + bn + bc); }
    __syncthreads();
    if (k0 + GT_K < K)
      __builtin_prefetch(A + (size_t)(bm + (tid>>2))*K + (k0 + GT_K), 0, 0);
    union { unsigned short u[16]; uint4 q[2]; v16bf v; } af, bf_;
    const int m0 = wm*16, n0 = wn*16;
    { const unsigned short* arow = &As[0][(m0+r)*GT_K];
      af.q[0] = *(const uint4*)&arow[8*hi];
      af.q[1] = *(const uint4*)&arow[16 + 8*hi]; }
    #pragma unroll
    for (int e = 0; e < 16; ++e)
      bf_.u[e] = Bs[0][(8*hi + (e&7) + 16*(e>>3))*GT_N + n0 + r];
    acc = __builtin_amdgcn_wmma_f32_16x16x32_bf16(false, af.v, false, bf_.v,
                                                  (short)0, acc, false, false);
  }
#endif

  const int col  = bn + wn*16 + r;
  const int rowb = bm + wm*16 + 8*hi;       // C/D layout: M = vgpr + 8*(lane>=16)
  if (col < N){
    #pragma unroll
    for (int i = 0; i < 8; ++i)
      if (rowb + i < M) Cz[(size_t)(rowb+i)*N + col] = acc[i];
  }
}

// ---------------- GEMV partial: sum over 128 of vec[d]*Wslice[d][jl] ------------
__device__ __forceinline__ float gemv_part(const float* __restrict__ vec512,
                                           const float* __restrict__ Wcol16,
                                           int jl, int p){
  float a = 0.f;
  #pragma unroll 8
  for (int d = p*128; d < (p+1)*128; ++d) a += vec512[d]*Wcol16[d*16 + jl];
  return a;
}

// ---------------- TTT sequential scan: 32 persistent WGs, 4 grid syncs/step ----
__global__ __launch_bounds__(256)
void ttt_scan(const float* __restrict__ kvq,    // (3, 4096, 512) = k,v,q
              const float* __restrict__ W1in,
              const float* __restrict__ W2in,
              float* __restrict__ W3g,          // (512,512) row-major, L2-resident
              float* __restrict__ hbuf,         // (4,512) scratch
              float* __restrict__ dbuf,         // (4,512) scratch
              float* __restrict__ hidden,       // (4096,512)
              unsigned* __restrict__ sync)
{
  __shared__ float W1s[512*16];   // owner-local column slice, LDS-resident
  __shared__ float W2s[512*16];
  __shared__ float W3s[512*16];   // LDS mirror of owned W3 columns
  __shared__ float ktL[4*512];
  __shared__ float qtL[4*512];
  __shared__ float hfull[4*512];
  __shared__ float dfull[4*512];
  __shared__ float part1[256], part2[256];
  __shared__ float z1L[64], z2L[64], dz1L[64], dz2L[64], doutL[64];

  const int g   = blockIdx.x;               // owns columns [g*16, g*16+16)
  const int tid = threadIdx.x;
  const int J0  = g*16;
  const int lane = tid & 31, wv = tid >> 5;
  const int lr = lane & 15, lh = lane >> 4;
  const int o  = tid >> 2, p = tid & 3;     // 4 partial threads per output
  const int ob = o >> 4, ojl = o & 15;

  const float* kmat = kvq;
  const float* vmat = kvq + (size_t)4096*512;
  const float* qmat = kvq + (size_t)2*4096*512;

  for (int idx = tid; idx < 512*16; idx += 256){
    int d = idx >> 4, jl = idx & 15;
    W1s[idx] = W1in[(size_t)d*512 + J0 + jl];
    W2s[idx] = W2in[(size_t)d*512 + J0 + jl];
    W3s[idx] = W3g [(size_t)d*512 + J0 + jl];
  }
  __syncthreads();

  for (int t = 0; t < 1024; ++t){
    // ---- stage kt, qt (4x512 f32 tiles, row stride N*H) via TDM ----
#if HAVE_TDM
    if (wv == 0){
      tdm_load(LDS_OFF(ktL), GA(kmat + (size_t)t*512), 512, 4, 512, 4,
               (unsigned long long)(1024*512), 2);
      tdm_load(LDS_OFF(qtL), GA(qmat + (size_t)t*512), 512, 4, 512, 4,
               (unsigned long long)(1024*512), 2);
      __builtin_amdgcn_s_wait_tensorcnt(0);
    }
    __syncthreads();
#else
    for (int idx = tid; idx < 2048; idx += 256){
      int b = idx >> 9, d = idx & 511;
      ktL[idx] = kmat[((size_t)b*1024 + t)*512 + d];
      qtL[idx] = qmat[((size_t)b*1024 + t)*512 + d];
    }
    __syncthreads();
#endif

    // ---- A: z1 = k@W1, z2 = k@W2, h = gelu(z1)*z2 on owned columns ----
    part1[tid] = gemv_part(&ktL[ob*512], W1s, ojl, p);
    part2[tid] = gemv_part(&ktL[ob*512], W2s, ojl, p);
    __syncthreads();
    if (tid < 64){
      float a1 = part1[tid*4]+part1[tid*4+1]+part1[tid*4+2]+part1[tid*4+3];
      float a2 = part2[tid*4]+part2[tid*4+1]+part2[tid*4+2]+part2[tid*4+3];
      z1L[tid] = a1; z2L[tid] = a2;
      hbuf[(size_t)(tid>>4)*512 + J0 + (tid&15)] = gelu_f(a1)*a2;
    }
    gsync(sync);

    // ---- B: out3 = h@W3 (owned cols), dout = (out3 - v)/B ----
#if HAVE_TDM
    if (wv == 0){
      tdm_load(LDS_OFF(hfull), GA(hbuf), 2048, 0, 2048, 0, 2048ull, 2);
      __builtin_amdgcn_s_wait_tensorcnt(0);
    }
    __syncthreads();
#else
    for (int idx = tid; idx < 2048; idx += 256) hfull[idx] = hbuf[idx];
    __syncthreads();
#endif
    part1[tid] = gemv_part(&hfull[ob*512], W3s, ojl, p);
    __syncthreads();
    if (tid < 64){
      float a = part1[tid*4]+part1[tid*4+1]+part1[tid*4+2]+part1[tid*4+3];
      int b = tid>>4, jl = tid&15;
      float dout = (a - vmat[((size_t)b*1024 + t)*512 + J0 + jl]) * 0.25f;
      doutL[tid] = dout;
      dbuf[(size_t)b*512 + J0 + jl] = dout;
    }
    gsync(sync);

    // ---- C: dh = dout@W3^T (rows from L2), dz1/dz2, rank-4 WMMA updates ----
#if HAVE_TDM
    if (wv == 0){
      tdm_load(LDS_OFF(dfull), GA(dbuf), 2048, 0, 2048, 0, 2048ull, 2);
      __builtin_amdgcn_s_wait_tensorcnt(0);
    }
    __syncthreads();
#else
    for (int idx = tid; idx < 2048; idx += 256) dfull[idx] = dbuf[idx];
    __syncthreads();
#endif
    {
      const float* w3row = W3g + (size_t)(J0 + ojl)*512;   // contiguous row read
      float a = 0.f;
      #pragma unroll 8
      for (int j2 = p*128; j2 < (p+1)*128; ++j2) a += dfull[ob*512 + j2]*w3row[j2];
      part1[tid] = a;
    }
    __syncthreads();
    if (tid < 64){
      float dh = part1[tid*4]+part1[tid*4+1]+part1[tid*4+2]+part1[tid*4+3];
      float z1 = z1L[tid], z2 = z2L[tid];
      dz1L[tid] = dh * z2 * gelu_grad_f(z1);
      dz2L[tid] = dh * gelu_f(z1);
    }
    __syncthreads();

    // W' = W + (-LR*a^T) x b : one V_WMMA_F32_16X16X4_F32 per 16x16 tile,
    // K = batch = 4, current W tile fed as the C operand.
    #pragma unroll
    for (int mt4 = 0; mt4 < 4; ++mt4){
      const int d0 = (wv*4 + mt4)*16;
      v2f a, bb; v8f c;
      // W1 <- W1 - LR * kt^T dz1
      a[0] = -LRATE*ktL[(2*lh+0)*512 + d0 + lr];
      a[1] = -LRATE*ktL[(2*lh+1)*512 + d0 + lr];
      bb[0] = dz1L[(2*lh+0)*16 + lr];
      bb[1] = dz1L[(2*lh+1)*16 + lr];
      #pragma unroll
      for (int i=0;i<8;++i) c[i] = W1s[(d0+i+8*lh)*16 + lr];
      c = __builtin_amdgcn_wmma_f32_16x16x4_f32(false,a,false,bb,(short)0,c,false,false);
      #pragma unroll
      for (int i=0;i<8;++i) W1s[(d0+i+8*lh)*16 + lr] = c[i];
      // W2 <- W2 - LR * kt^T dz2
      bb[0] = dz2L[(2*lh+0)*16 + lr];
      bb[1] = dz2L[(2*lh+1)*16 + lr];
      #pragma unroll
      for (int i=0;i<8;++i) c[i] = W2s[(d0+i+8*lh)*16 + lr];
      c = __builtin_amdgcn_wmma_f32_16x16x4_f32(false,a,false,bb,(short)0,c,false,false);
      #pragma unroll
      for (int i=0;i<8;++i) W2s[(d0+i+8*lh)*16 + lr] = c[i];
      // W3 <- W3 - LR * h^T dout
      a[0] = -LRATE*hfull[(2*lh+0)*512 + d0 + lr];
      a[1] = -LRATE*hfull[(2*lh+1)*512 + d0 + lr];
      bb[0] = doutL[(2*lh+0)*16 + lr];
      bb[1] = doutL[(2*lh+1)*16 + lr];
      #pragma unroll
      for (int i=0;i<8;++i) c[i] = W3s[(d0+i+8*lh)*16 + lr];
      c = __builtin_amdgcn_wmma_f32_16x16x4_f32(false,a,false,bb,(short)0,c,false,false);
      #pragma unroll
      for (int i=0;i<8;++i) W3s[(d0+i+8*lh)*16 + lr] = c[i];
    }
    __syncthreads();
    gsync(sync);   // all WGs done reading *old* W3 rows -> safe to write back

    // ---- write back W3 slice (TDM store), forward at q with updated weights ----
#if HAVE_TDM && HAVE_TDM_ST
    if (wv == 0)
      tdm_store(LDS_OFF(W3s), GA(W3g + J0), 16, 512, 512, 512, 512ull, 2);
#else
    for (int idx = tid; idx < 512*16; idx += 256){
      int d = idx >> 4, jl = idx & 15;
      W3g[(size_t)d*512 + J0 + jl] = W3s[idx];
    }
#endif
    part1[tid] = gemv_part(&qtL[ob*512], W1s, ojl, p);
    part2[tid] = gemv_part(&qtL[ob*512], W2s, ojl, p);
    __syncthreads();
    if (tid < 64){
      float a1 = part1[tid*4]+part1[tid*4+1]+part1[tid*4+2]+part1[tid*4+3];
      float a2 = part2[tid*4]+part2[tid*4+1]+part2[tid*4+2]+part2[tid*4+3];
      dbuf[(size_t)(tid>>4)*512 + J0 + (tid&15)] = gelu_f(a1)*a2;  // hq
    }
#if HAVE_TDM && HAVE_TDM_ST
    if (wv == 0) __builtin_amdgcn_s_wait_tensorcnt(0);  // W3 store globally done
#endif
    gsync(sync);

    // ---- D: hidden_t = hq @ W3' on owned columns ----
#if HAVE_TDM
    if (wv == 0){
      tdm_load(LDS_OFF(hfull), GA(dbuf), 2048, 0, 2048, 0, 2048ull, 2);
      __builtin_amdgcn_s_wait_tensorcnt(0);
    }
    __syncthreads();
#else
    for (int idx = tid; idx < 2048; idx += 256) hfull[idx] = dbuf[idx];
    __syncthreads();
#endif
    part1[tid] = gemv_part(&hfull[ob*512], W3s, ojl, p);
    __syncthreads();
    if (tid < 64){
      float a = part1[tid*4]+part1[tid*4+1]+part1[tid*4+2]+part1[tid*4+3];
      int b = tid>>4, jl = tid&15;
      hidden[((size_t)b*1024 + t)*512 + J0 + jl] = a;
    }
    __syncthreads();
  }
}

// ---------------- RMSNorm + cast to bf16 ----------------
__global__ __launch_bounds__(256)
void rmsnorm_cast(const float* __restrict__ hid, const float* __restrict__ scale,
                  __hip_bfloat16* __restrict__ out){
  __shared__ float red[8];
  const int row = blockIdx.x;                 // 4096 rows of H=512
  const float* h = hid + (size_t)row*512;
  float s = 0.f;
  for (int j = threadIdx.x; j < 512; j += 256){ float v = h[j]; s += v*v; }
  for (int off = 16; off; off >>= 1) s += __shfl_down(s, off, 32);
  if ((threadIdx.x & 31) == 0) red[threadIdx.x >> 5] = s;
  __syncthreads();
  if (threadIdx.x == 0){
    float t = 0.f;
    #pragma unroll
    for (int i = 0; i < 8; ++i) t += red[i];
    red[0] = rsqrtf(t*(1.0f/512.0f) + 1e-6f);
  }
  __syncthreads();
  float rn = red[0];
  for (int j = threadIdx.x; j < 512; j += 256)
    out[(size_t)row*512 + j] = __float2bfloat16(h[j]*rn*scale[j]);
}

// ---------------- launch ----------------
extern "C" void kernel_launch(void* const* d_in, const int* in_sizes, int n_in,
                              void* d_out, int out_size, void* d_ws, size_t ws_size,
                              hipStream_t stream){
  (void)in_sizes; (void)n_in; (void)out_size; (void)ws_size;
  const float* x    = (const float*)d_in[0];
  const float* Wkvq = (const float*)d_in[1];
  const float* W1   = (const float*)d_in[2];
  const float* W2   = (const float*)d_in[3];
  const float* W3   = (const float*)d_in[4];
  const float* nsc  = (const float*)d_in[5];
  const float* Wout = (const float*)d_in[6];
  float* out = (float*)d_out;

  char* ws = (char*)d_ws;                       // ~49 MB used
  __hip_bfloat16* xbf    = (__hip_bfloat16*)(ws + 0);          //  8 MB
  __hip_bfloat16* wkvqbf = (__hip_bfloat16*)(ws + 8388608);    //  3 MB
  __hip_bfloat16* woutbf = (__hip_bfloat16*)(ws + 11534336);   //  1 MB
  float*          kvqf   = (float*)         (ws + 12582912);   // 24 MB
  float*          hid    = (float*)         (ws + 37748736);   //  8 MB
  __hip_bfloat16* hidbf  = (__hip_bfloat16*)(ws + 46137344);   //  4 MB
  float*          W3g    = (float*)         (ws + 50331648);   //  1 MB
  float*          hbuf   = (float*)         (ws + 51380224);
  float*          dbuf   = (float*)         (ws + 51388416);
  unsigned*       syn    = (unsigned*)      (ws + 51396608);

  cast_f32_to_bf16<<<(4194304+255)/256, 256, 0, stream>>>(x,    xbf,    4194304);
  cast_f32_to_bf16<<<(1572864+255)/256, 256, 0, stream>>>(Wkvq, wkvqbf, 1572864);
  cast_f32_to_bf16<<<( 524288+255)/256, 256, 0, stream>>>(Wout, woutbf,  524288);
  init_state<<<256, 256, 0, stream>>>(W3, W3g, syn);

  dim3 gproj(4096/GT_M, 512/GT_N, 3);    // k,v,q projections
  gemm_bf16_wmma<<<gproj, 256, 0, stream>>>(xbf, wkvqbf, kvqf,
      4096, 512, 1024, (size_t)1024*512, (size_t)4096*512);

  ttt_scan<<<NWG, 256, 0, stream>>>(kvqf, W1, W2, W3g, hbuf, dbuf, hid, syn);

  rmsnorm_cast<<<4096, 256, 0, stream>>>(hid, nsc, hidbf);

  dim3 gout(4096/GT_M, 1024/GT_N, 1);    // out projection
  gemm_bf16_wmma<<<gout, 256, 0, stream>>>(hidbf, woutbf, out,
      4096, 1024, 512, 0, 0);
}
